// Self_Attention_30769145709333
// MI455X (gfx1250) — matrix operand order repaired and checked by reading, not compiled, not run on hardware
//
#include <hip/hip_runtime.h>
#include <hip/hip_bf16.h>

typedef __attribute__((ext_vector_type(16))) __bf16 v16bf;
typedef __attribute__((ext_vector_type(8)))  __bf16 v8bf;
typedef __attribute__((ext_vector_type(8)))  float  v8f;

#define NHEADS 16
#define HD 64

__device__ __forceinline__ v8f wmma_bf16(v16bf a, v16bf b, v8f c) {
    return __builtin_amdgcn_wmma_f32_16x16x32_bf16(false, a, false, b, (short)0, c,
                                                   false, false);
}

// ---------------------------------------------------------------------------
// fp32 -> bf16 bulk convert (n must be a multiple of 4; all our sizes are)
// ---------------------------------------------------------------------------
__global__ void f32_to_bf16_kernel(const float* __restrict__ src,
                                   __bf16* __restrict__ dst, long n) {
    long i = ((long)blockIdx.x * blockDim.x + threadIdx.x) * 4;
    if (i + 3 >= n) return;
    float4 v = *(const float4*)(src + i);
    dst[i + 0] = (__bf16)v.x;
    dst[i + 1] = (__bf16)v.y;
    dst[i + 2] = (__bf16)v.z;
    dst[i + 3] = (__bf16)v.w;
}

// ---------------------------------------------------------------------------
// C[M,N] = A[M,K](bf16,row-major) * W[N,K](bf16,row-major)^T + bias (f32)
// One wave computes a 32x64 tile (2x4 grid of 16x16 WMMA tiles): 8 WMMAs and
// 12 b128 loads per K-step of 32.  8 waves / block.
// ---------------------------------------------------------------------------
template <bool OUT_BF16>
__global__ __launch_bounds__(256)
void gemm_bias_kernel(const __bf16* __restrict__ A, const __bf16* __restrict__ W,
                      const float* __restrict__ bias, void* __restrict__ Cout,
                      int M, int N, int K) {
    const int wave = threadIdx.x >> 5;
    const int lane = threadIdx.x & 31;
    const int ntn  = N >> 6;                       // 64-wide column tiles
    const int tile = blockIdx.x * 8 + wave;
    if (tile >= (M >> 5) * ntn) return;
    const int m0 = (tile / ntn) << 5;
    const int n0 = (tile % ntn) << 6;

    const int hv  = lane >> 4;
    const int l15 = lane & 15;

    // A fragment rows (two 16-row tiles)
    const __bf16* aptr0 = A + (long)(m0 + l15) * K + hv * 16;
    const __bf16* aptr1 = A + (long)(m0 + 16 + l15) * K + hv * 16;
    // B fragment rows (four 16-col tiles), 16 contiguous K per lane
    const __bf16* wptr0 = W + (long)(n0 + l15) * K + hv * 16;
    const __bf16* wptr1 = W + (long)(n0 + 16 + l15) * K + hv * 16;
    const __bf16* wptr2 = W + (long)(n0 + 32 + l15) * K + hv * 16;
    const __bf16* wptr3 = W + (long)(n0 + 48 + l15) * K + hv * 16;

    v8f acc[2][4] = {};
    for (int k0 = 0; k0 < K; k0 += 32) {
        __builtin_prefetch(aptr0 + k0 + 256, 0, 1);
        __builtin_prefetch(wptr0 + k0 + 256, 0, 1);
        union { v16bf v; v8bf h[2]; } af0, af1;
        af0.h[0] = *(const v8bf*)(aptr0 + k0);
        af0.h[1] = *(const v8bf*)(aptr0 + k0 + 8);
        af1.h[0] = *(const v8bf*)(aptr1 + k0);
        af1.h[1] = *(const v8bf*)(aptr1 + k0 + 8);
        v16bf b0 = *(const v16bf*)(wptr0 + k0);
        v16bf b1 = *(const v16bf*)(wptr1 + k0);
        v16bf b2 = *(const v16bf*)(wptr2 + k0);
        v16bf b3 = *(const v16bf*)(wptr3 + k0);
        acc[0][0] = wmma_bf16(af0.v, b0, acc[0][0]);
        acc[0][1] = wmma_bf16(af0.v, b1, acc[0][1]);
        acc[0][2] = wmma_bf16(af0.v, b2, acc[0][2]);
        acc[0][3] = wmma_bf16(af0.v, b3, acc[0][3]);
        acc[1][0] = wmma_bf16(af1.v, b0, acc[1][0]);
        acc[1][1] = wmma_bf16(af1.v, b1, acc[1][1]);
        acc[1][2] = wmma_bf16(af1.v, b2, acc[1][2]);
        acc[1][3] = wmma_bf16(af1.v, b3, acc[1][3]);
    }

#pragma unroll
    for (int ni = 0; ni < 4; ++ni) {
        const int   col = n0 + ni * 16 + l15;      // lane's C column
        const float bv  = bias[col];
#pragma unroll
        for (int mi = 0; mi < 2; ++mi) {
#pragma unroll
            for (int r = 0; r < 8; ++r) {
                const int   row = m0 + mi * 16 + r + 8 * hv;
                const float v   = acc[mi][ni][r] + bv;
                if (OUT_BF16)
                    ((__bf16*)Cout)[(long)row * N + col] = (__bf16)v;
                else
                    ((float*)Cout)[(long)row * N + col] = v;
            }
        }
    }
}

// ---------------------------------------------------------------------------
// Flash attention over bf16 qkv [B*S, 3*E] (layout: h*192 + {q:0..63,k:64..127,
// v:128..191}).  One wave per (b, h, 16-row q-tile); 64 keys per iteration:
// 16 WMMAs (8 Q.K^T + 8 P.V) per softmax round.  V staging is issued at the
// top of the iteration and all K fragments are loaded in one batch so the
// vmem latency overlaps the WMMA chain and softmax VALU work.
// ---------------------------------------------------------------------------
__global__ __launch_bounds__(32)
void flash_attn_kernel(const __bf16* __restrict__ qkv, __bf16* __restrict__ attn,
                       int S) {
    const int QT  = S >> 4;
    const int qt  = blockIdx.x % QT;
    const int h   = (blockIdx.x / QT) % NHEADS;
    const int b   = blockIdx.x / (QT * NHEADS);
    const int lane = threadIdx.x;
    const int hv   = lane >> 4;
    const int l15  = lane & 15;
    const int E3   = 3 * NHEADS * HD;           // 3072
    const int hb   = h * (3 * HD);              // head base column
    const int q0   = qt << 4;
    const __bf16* base = qkv + (long)b * S * E3;

    __shared__ __bf16 Vt[64 * HD];   // 8 KB staged V tile (kpos-major)
    __shared__ __bf16 Pt[16 * 64];   // 2 KB probability tile (row-major)

    // Q as A fragments (rows q0..q0+15, d split into 0..31 / 32..63)
    const __bf16* qrow = base + (long)(q0 + l15) * E3 + hb;
    union { v16bf v; v8bf h2[2]; } qa0, qa1;
    qa0.h2[0] = *(const v8bf*)(qrow + hv * 16);
    qa0.h2[1] = *(const v8bf*)(qrow + hv * 16 + 8);
    qa1.h2[0] = *(const v8bf*)(qrow + 32 + hv * 16);
    qa1.h2[1] = *(const v8bf*)(qrow + 32 + hv * 16 + 8);

    v8f o[4] = {};
    float rmax[8], rsum[8];
#pragma unroll
    for (int r = 0; r < 8; ++r) { rmax[r] = -1e30f; rsum[r] = 0.f; }
    const float scale = 0.125f;              // 1/sqrt(64)

    const int jmax = (q0 + 15) >> 6;         // inclusive: last 64-key tile
    for (int j = 0; j <= jmax; ++j) {
        const int kb = j << 6;

        // --- stage V tile [64 kpos x 64 d] into LDS first: these loads and
        //     ds stores drain in the shadow of the score WMMAs + softmax ---
#pragma unroll
        for (int rr = 0; rr < 2; ++rr) {
            const int kpos = rr * 32 + lane;
            const __bf16* vrow = base + (long)(kb + kpos) * E3 + hb + 2 * HD;
            const v8bf* src = (const v8bf*)vrow;
            v8bf*       dst = (v8bf*)(Vt + kpos * HD);
#pragma unroll
            for (int t = 0; t < 8; ++t) dst[t] = src[t];
        }

        // --- batch-load all 8 K fragments (one clause), then 8 WMMAs ---
        v16bf klo[4], khi[4];
#pragma unroll
        for (int t = 0; t < 4; ++t) {
            const __bf16* kr = base + (long)(kb + t * 16 + l15) * E3 + hb + HD;
            klo[t] = *(const v16bf*)(kr + hv * 16);
            khi[t] = *(const v16bf*)(kr + 32 + hv * 16);
        }
        v8f s[4] = {};
#pragma unroll
        for (int t = 0; t < 4; ++t) s[t] = wmma_bf16(qa0.v, klo[t], s[t]);
#pragma unroll
        for (int t = 0; t < 4; ++t) s[t] = wmma_bf16(qa1.v, khi[t], s[t]);

        // --- online softmax (row m = r + 8*hv; col n = kb + 16t + l15) ---
#pragma unroll
        for (int r = 0; r < 8; ++r) {
            const int qg = q0 + r + 8 * hv;
            float a[4];
#pragma unroll
            for (int t = 0; t < 4; ++t) {
                float v = s[t][r] * scale;
                a[t] = (kb + t * 16 + l15 > qg) ? -1e30f : v;   // causal mask
            }
            float mx = fmaxf(fmaxf(a[0], a[1]), fmaxf(a[2], a[3]));
#pragma unroll
            for (int off = 8; off; off >>= 1)
                mx = fmaxf(mx, __shfl_xor(mx, off, 16));
            const float mnew  = fmaxf(rmax[r], mx);
            const float alpha = __expf(rmax[r] - mnew);
            rmax[r] = mnew;
            float p[4], ps = 0.f;
#pragma unroll
            for (int t = 0; t < 4; ++t) { p[t] = __expf(a[t] - mnew); ps += p[t]; }
#pragma unroll
            for (int off = 8; off; off >>= 1)
                ps += __shfl_xor(ps, off, 16);
            rsum[r] = rsum[r] * alpha + ps;
            o[0][r] *= alpha; o[1][r] *= alpha; o[2][r] *= alpha; o[3][r] *= alpha;
            const int m = r + 8 * hv;
#pragma unroll
            for (int t = 0; t < 4; ++t)
                Pt[m * 64 + t * 16 + l15] = (__bf16)p[t];
        }

        __syncthreads();   // single-wave WG: lowers to dscnt waits

        // P as two A fragments (K = kpos 0..31 and 32..63)
        union { v16bf v; v8bf h2[2]; } pa0, pa1;
        pa0.h2[0] = *(const v8bf*)(Pt + l15 * 64 + hv * 16);
        pa0.h2[1] = *(const v8bf*)(Pt + l15 * 64 + hv * 16 + 8);
        pa1.h2[0] = *(const v8bf*)(Pt + l15 * 64 + 32 + hv * 16);
        pa1.h2[1] = *(const v8bf*)(Pt + l15 * 64 + 32 + hv * 16 + 8);

        // V^T as B fragments from LDS: n = d, K = kpos (strided gather)
#pragma unroll
        for (int t = 0; t < 4; ++t) {
            v16bf vlo, vhi;
#pragma unroll
            for (int e = 0; e < 16; ++e) {
                const int col = t * 16 + l15;
                vlo[e] = Vt[(hv * 16 + e) * HD + col];
                vhi[e] = Vt[(32 + hv * 16 + e) * HD + col];
            }
            o[t] = wmma_bf16(pa0.v, vlo, o[t]);
            o[t] = wmma_bf16(pa1.v, vhi, o[t]);
        }
        __syncthreads();
    }

    // epilogue: normalize and store bf16 attn [B*S, E] at column h*64 + d
#pragma unroll
    for (int r = 0; r < 8; ++r) {
        const float inv = 1.0f / rsum[r];
        const int   row = q0 + r + 8 * hv;
        const long  off = ((long)b * S + row) * (NHEADS * HD) + h * HD + l15;
#pragma unroll
        for (int t = 0; t < 4; ++t)
            attn[off + 16 * t] = (__bf16)(o[t][r] * inv);
    }
}

// ---------------------------------------------------------------------------
extern "C" void kernel_launch(void* const* d_in, const int* in_sizes, int n_in,
                              void* d_out, int out_size, void* d_ws, size_t ws_size,
                              hipStream_t stream) {
    const float* x    = (const float*)d_in[0];
    const float* Wqkv = (const float*)d_in[1];
    const float* bqkv = (const float*)d_in[2];
    const float* Wout = (const float*)d_in[3];
    const float* bout = (const float*)d_in[4];
    float* out = (float*)d_out;

    const int  B = 4, S = 2048, E = 1024;
    const long M = (long)B * S;                 // 8192

    char* ws = (char*)d_ws;
    __bf16* x_bf    = (__bf16*)(ws);                          // 16 MB
    __bf16* wqkv_bf = (__bf16*)(ws + (16L << 20));            //  6 MB
    __bf16* wout_bf = (__bf16*)(ws + (24L << 20));            //  2 MB
    __bf16* qkv_bf  = (__bf16*)(ws + (32L << 20));            // 48 MB
    __bf16* attn_bf = (__bf16*)(ws + (80L << 20));            // 16 MB

    // fp32 -> bf16 converts
    {
        long n;
        n = M * E;
        f32_to_bf16_kernel<<<(int)(n / 4 / 256), 256, 0, stream>>>(x, x_bf, n);
        n = 3L * E * E;
        f32_to_bf16_kernel<<<(int)(n / 4 / 256), 256, 0, stream>>>(Wqkv, wqkv_bf, n);
        n = (long)E * E;
        f32_to_bf16_kernel<<<(int)(n / 4 / 256), 256, 0, stream>>>(Wout, wout_bf, n);
    }

    // QKV projection: [8192,1024] x [3072,1024]^T -> bf16 [8192,3072]
    {
        const int tiles = (int)(M >> 5) * ((3 * E) >> 6);     // 256*48
        gemm_bias_kernel<true><<<(tiles + 7) / 8, 256, 0, stream>>>(
            x_bf, wqkv_bf, bqkv, qkv_bf, (int)M, 3 * E, E);
    }

    // Causal flash attention -> bf16 attn [8192,1024]
    flash_attn_kernel<<<B * NHEADS * (S >> 4), 32, 0, stream>>>(qkv_bf, attn_bf, S);

    // Output projection: [8192,1024] x [1024,1024]^T -> f32 d_out
    {
        const int tiles = (int)(M >> 5) * (E >> 6);           // 256*16
        gemm_bias_kernel<false><<<(tiles + 7) / 8, 256, 0, stream>>>(
            attn_bf, wout_bf, bout, out, (int)M, E, E);
    }
}
